// SimpleAttentionTextClassifier_14173392076803
// MI455X (gfx1250) — matrix operand order repaired and checked by reading, not compiled, not run on hardware
//
#include <hip/hip_runtime.h>
#include <hip/hip_bf16.h>

typedef __attribute__((ext_vector_type(16))) __bf16 v16bf;
typedef __attribute__((ext_vector_type(8)))  float  v8f;

#define EMBED_DIM  256
#define NCLS       2
#define BATCH      16
#define SEQ        2048
#define ROWS_PER_BLOCK 128
#define WAVES      8
#define KCHUNK     64
#define NCHUNK     (SEQ / KCHUNK)
#define SK_STRIDE  264   // 528B rows (16B multiple)
#define SV_STRIDE  72    // 144B rows (16B multiple)
#define SP_STRIDE  64    // 128B rows (16B multiple)
#define TOK_TILE   64
#define ST_STRIDE  264

union Frag {
    v16bf        v;
    uint4        q[2];
    unsigned int u[8];
};

// ---- CDNA5 async global->LDS copy (ASYNCcnt-tracked, no VGPR staging) ----
__device__ __forceinline__ void async_copy_b128(void* lds_ptr, const void* gptr) {
    unsigned int lo = (unsigned int)(uintptr_t)lds_ptr;   // LDS offset = addr[31:0]
    unsigned long long ga = (unsigned long long)(uintptr_t)gptr;
    asm volatile("global_load_async_to_lds_b128 %0, %1, off"
                 :: "v"(lo), "v"(ga) : "memory");
}
__device__ __forceinline__ void wait_async0() {
    asm volatile("s_wait_asynccnt 0" ::: "memory");
}

// ---------------------------------------------------------------------------
// Kernel 0: zero the pooled accumulator
// ---------------------------------------------------------------------------
__global__ void zero_pooled(float* __restrict__ pooled) {
    int i = blockIdx.x * blockDim.x + threadIdx.x;
    if (i < BATCH * EMBED_DIM) pooled[i] = 0.0f;
}

// ---------------------------------------------------------------------------
// Kernel 1: embedding gather -> bf16; writes row-major emb[b][s][d] and
// column-major embT[b][d][s] (transpose staged once through LDS).
// ---------------------------------------------------------------------------
__global__ __launch_bounds__(256)
void gather_embed(const int* __restrict__ ids, const float* __restrict__ table,
                  __bf16* __restrict__ emb, __bf16* __restrict__ embT) {
    __shared__ __align__(16) __bf16 sT[TOK_TILE][ST_STRIDE];
    const int b     = blockIdx.y;
    const int stile = blockIdx.x;
    const int tid   = threadIdx.x;
    {
        const int token = tid >> 2;
        const int dg    = tid & 3;
        const int s     = stile * TOK_TILE + token;
        const int row   = ids[b * SEQ + s];
        const float4* src = (const float4*)(table + (size_t)row * EMBED_DIM + dg * 64);
        __bf16* dst = emb + ((size_t)(b * SEQ + s)) * EMBED_DIM + dg * 64;
#pragma unroll
        for (int j = 0; j < 8; ++j) {
            float4 f0 = src[2 * j];
            float4 f1 = src[2 * j + 1];
            union { __bf16 h[8]; uint4 u; } o;
            o.h[0] = (__bf16)f0.x; o.h[1] = (__bf16)f0.y;
            o.h[2] = (__bf16)f0.z; o.h[3] = (__bf16)f0.w;
            o.h[4] = (__bf16)f1.x; o.h[5] = (__bf16)f1.y;
            o.h[6] = (__bf16)f1.z; o.h[7] = (__bf16)f1.w;
            *(uint4*)(dst + j * 8) = o.u;
            *(uint4*)&sT[token][dg * 64 + j * 8] = o.u;
        }
    }
    __syncthreads();
    {
        const int d = tid;
        union { __bf16 h[64]; uint4 u[8]; } o;
#pragma unroll
        for (int k = 0; k < TOK_TILE; ++k) o.h[k] = sT[k][d];
        uint4* dst = (uint4*)(embT + ((size_t)(b * EMBED_DIM + d)) * SEQ + stile * TOK_TILE);
#pragma unroll
        for (int j = 0; j < 8; ++j) dst[j] = o.u[j];
    }
}

// ---------------------------------------------------------------------------
// Kernel 2: fused flash attention + pooling, async double-buffered K/V in LDS
// ---------------------------------------------------------------------------
__global__ __launch_bounds__(256, 1)
void attn_kernel(const __bf16* __restrict__ emb, const __bf16* __restrict__ embT,
                 float* __restrict__ pooled) {
    __shared__ __align__(16) __bf16 sK[2][KCHUNK][SK_STRIDE];     // K rows (GEMM-1 B)
    __shared__ __align__(16) __bf16 sV[2][EMBED_DIM][SV_STRIDE];  // embT rows (GEMM-2 B)
    __shared__ __align__(16) __bf16 sP[WAVES][16][SP_STRIDE];     // per-wave P staging

    const int batch   = blockIdx.y;
    const int rowTile = blockIdx.x;
    const int tid     = threadIdx.x;
    const int wave    = tid >> 5;
    const int lane    = tid & 31;
    const int m       = lane & 15;
    const int half    = lane >> 4;
    const int rowBase = rowTile * ROWS_PER_BLOCK + wave * 16;
    const __bf16* embB  = emb  + (size_t)batch * SEQ * EMBED_DIM;
    const __bf16* embTB = embT + (size_t)batch * EMBED_DIM * SEQ;
    const __bf16* qrow  = embB + (size_t)(rowBase + m) * EMBED_DIM;

    // cooperative async copy of one 64-key chunk into buffer `buf`
    auto issue_chunk = [&](int buf, int key0) {
#pragma unroll
        for (int i = 0; i < 8; ++i) {           // sK: 64 rows x 512B
            int slice = i * 256 + tid;
            int row = slice >> 5, c = slice & 31;
            async_copy_b128(&sK[buf][row][c * 8],
                            embB + (size_t)(key0 + row) * EMBED_DIM + c * 8);
        }
#pragma unroll
        for (int i = 0; i < 8; ++i) {           // sV: 256 rows x 128B
            int slice = i * 256 + tid;
            int row = slice >> 3, c = slice & 7;
            async_copy_b128(&sV[buf][row][c * 8],
                            embTB + (size_t)row * SEQ + key0 + c * 8);
        }
    };

    const v8f vzero = {0.f, 0.f, 0.f, 0.f, 0.f, 0.f, 0.f, 0.f};
    const v8f l2e   = {1.4426950408889634f, 1.4426950408889634f, 1.4426950408889634f,
                       1.4426950408889634f, 1.4426950408889634f, 1.4426950408889634f,
                       1.4426950408889634f, 1.4426950408889634f};
    v8f acc[16];
#pragma unroll
    for (int t = 0; t < 16; ++t) acc[t] = vzero;
    float mrow[8], lrow[8];
#pragma unroll
    for (int r = 0; r < 8; ++r) { mrow[r] = -1e30f; lrow[r] = 0.0f; }

    issue_chunk(0, 0);
    wait_async0();
    __syncthreads();

    for (int kc = 0; kc < NCHUNK; ++kc) {
        const int buf  = kc & 1;
        const int key0 = kc * KCHUNK;
        if (kc + 1 < NCHUNK) issue_chunk(buf ^ 1, key0 + KCHUNK);

        // ---- S = Q . K^T : 4 key n-tiles x 8 d-chunks = 32 WMMAs ----
        v8f s[4] = {vzero, vzero, vzero, vzero};
#pragma unroll
        for (int dc = 0; dc < 8; ++dc) {
            Frag a;
            a.q[0] = *(const uint4*)(qrow + dc * 32 + half * 16);
            a.q[1] = *(const uint4*)(qrow + dc * 32 + half * 16 + 8);
#pragma unroll
            for (int nt = 0; nt < 4; ++nt) {
                Frag bk;
                const uint4* kr = (const uint4*)&sK[buf][nt * 16 + m][dc * 32 + half * 16];
                bk.q[0] = kr[0];
                bk.q[1] = kr[1];
                s[nt] = __builtin_amdgcn_wmma_f32_16x16x32_bf16(
                    false, a.v, false, bk.v, (short)0, s[nt], false, false);
            }
        }

        // ---- online softmax in log2 domain ----
#pragma unroll
        for (int nt = 0; nt < 4; ++nt) s[nt] *= l2e;
        v8f scv;
#pragma unroll
        for (int r = 0; r < 8; ++r) {
            float tmax = fmaxf(fmaxf(s[0][r], s[1][r]), fmaxf(s[2][r], s[3][r]));
#pragma unroll
            for (int off = 8; off >= 1; off >>= 1)
                tmax = fmaxf(tmax, __shfl_xor(tmax, off, 16));
            float mn = fmaxf(mrow[r], tmax);
            float sc = __builtin_amdgcn_exp2f(mrow[r] - mn);
            float p0 = __builtin_amdgcn_exp2f(s[0][r] - mn);
            float p1 = __builtin_amdgcn_exp2f(s[1][r] - mn);
            float p2 = __builtin_amdgcn_exp2f(s[2][r] - mn);
            float p3 = __builtin_amdgcn_exp2f(s[3][r] - mn);
            float rs = (p0 + p1) + (p2 + p3);
#pragma unroll
            for (int off = 8; off >= 1; off >>= 1)
                rs += __shfl_xor(rs, off, 16);
            lrow[r] = lrow[r] * sc + rs;
            mrow[r] = mn;
            s[0][r] = p0; s[1][r] = p1; s[2][r] = p2; s[3][r] = p3;
            scv[r]  = sc;
        }
#pragma unroll
        for (int t = 0; t < 16; ++t) acc[t] *= scv;

        // ---- stage P (C layout f32 -> A layout bf16) through LDS ----
#pragma unroll
        for (int r = 0; r < 8; ++r)
#pragma unroll
            for (int nt = 0; nt < 4; ++nt)
                sP[wave][r + half * 8][nt * 16 + m] = (__bf16)s[nt][r];

        // ---- ctx += P (16x64) . V (64x256): 2 k-chunks x 16 n-tiles ----
#pragma unroll
        for (int kk = 0; kk < 2; ++kk) {
            Frag p;
            p.q[0] = *(const uint4*)&sP[wave][m][kk * 32 + half * 16];
            p.q[1] = *(const uint4*)&sP[wave][m][kk * 32 + half * 16 + 8];
#pragma unroll
            for (int t = 0; t < 16; ++t) {
                Frag bv;
                const uint4* vr = (const uint4*)&sV[buf][t * 16 + m][kk * 32 + half * 16];
                bv.q[0] = vr[0];
                bv.q[1] = vr[1];
                acc[t] = __builtin_amdgcn_wmma_f32_16x16x32_bf16(
                    false, p.v, false, bv.v, (short)0, acc[t], false, false);
            }
        }

        wait_async0();     // next chunk landed in LDS
        __syncthreads();   // all waves done with current buffer + copies visible
    }

    // ---- normalize rows by 1/l, sum the 16 rows, accumulate into pooled ----
    v8f invv;
#pragma unroll
    for (int r = 0; r < 8; ++r) invv[r] = 1.0f / lrow[r];
#pragma unroll
    for (int t = 0; t < 16; ++t) {
        v8f a = acc[t] * invv;
        float cs = 0.0f;
#pragma unroll
        for (int r = 0; r < 8; ++r) cs += a[r];
        cs += __shfl_xor(cs, 16, 32);
        if (half == 0)
            atomicAdd(&pooled[batch * EMBED_DIM + t * 16 + m], cs);
    }
}

// ---------------------------------------------------------------------------
// Kernel 3: classifier head
// ---------------------------------------------------------------------------
__global__ void head_kernel(const float* __restrict__ pooled,
                            const float* __restrict__ W,
                            const float* __restrict__ bias,
                            float* __restrict__ out) {
    int t = threadIdx.x;
    if (t < BATCH * NCLS) {
        int bi = t >> 1, c = t & 1;
        const float* p = pooled + bi * EMBED_DIM;
        const float* w = W + c * EMBED_DIM;
        float s = 0.0f;
        for (int d = 0; d < EMBED_DIM; ++d) s += p[d] * w[d];
        out[t] = s * (1.0f / SEQ) + bias[c];
    }
}

// ---------------------------------------------------------------------------
extern "C" void kernel_launch(void* const* d_in, const int* in_sizes, int n_in,
                              void* d_out, int out_size, void* d_ws, size_t ws_size,
                              hipStream_t stream) {
    (void)in_sizes; (void)n_in; (void)out_size; (void)ws_size;
    const int*   ids   = (const int*)d_in[0];
    const float* table = (const float*)d_in[1];
    const float* W     = (const float*)d_in[2];
    const float* bias  = (const float*)d_in[3];
    float*       out   = (float*)d_out;

    const size_t embBytes = (size_t)BATCH * SEQ * EMBED_DIM * sizeof(__bf16);
    __bf16* emb    = (__bf16*)d_ws;
    __bf16* embT   = (__bf16*)((char*)d_ws + embBytes);
    float*  pooled = (float*)((char*)d_ws + 2 * embBytes);

    zero_pooled<<<(BATCH * EMBED_DIM + 255) / 256, 256, 0, stream>>>(pooled);

    dim3 ggrid(SEQ / TOK_TILE, BATCH);
    gather_embed<<<ggrid, 256, 0, stream>>>(ids, table, emb, embT);

    dim3 agrid(SEQ / ROWS_PER_BLOCK, BATCH);
    attn_kernel<<<agrid, 256, 0, stream>>>(emb, embT, pooled);

    head_kernel<<<1, 64, 0, stream>>>(pooled, W, bias, out);
}
